// SandwichLayer_11433202942560
// MI455X (gfx1250) — matrix operand order repaired and checked by reading, not compiled, not run on hardware
//
#include <hip/hip_runtime.h>
#include <hip/hip_bf16.h>
#include <stdint.h>

#define NEG_SLOPE 0.2f
#define BN_EPS 1e-5f

typedef __attribute__((ext_vector_type(16))) __bf16 v16bf;
typedef __attribute__((ext_vector_type(8)))  float  v8f;
typedef __attribute__((ext_vector_type(4)))  unsigned int v4u;
typedef __attribute__((ext_vector_type(8)))  int v8i_t;
typedef __attribute__((ext_vector_type(4)))  int v4i_t;

static __device__ __forceinline__ float sigmoidf_(float x) { return 1.0f / (1.0f + __expf(-x)); }
static __device__ __forceinline__ float reluf_(float x) { return x > 0.0f ? x : 0.0f; }
static __device__ __forceinline__ unsigned short f2bf(float f) {
  unsigned u = __float_as_uint(f);
  u += 0x7FFFu + ((u >> 16) & 1u);          // round-to-nearest-even bf16
  return (unsigned short)(u >> 16);
}
// order-preserving float<->uint map for atomicMax on floats (0 == very negative)
static __device__ __forceinline__ unsigned fenc(float f) {
  unsigned b = __float_as_uint(f);
  return (b & 0x80000000u) ? ~b : (b | 0x80000000u);
}
static __device__ __forceinline__ float fdec(unsigned u) {
  return (u & 0x80000000u) ? __uint_as_float(u & 0x7FFFFFFFu) : __uint_as_float(~u);
}

// ---------------- K1: residual conv1x1 + gated_conv1 (k=3, C=32,T=8) + bn0 stats ----------------
__global__ void __launch_bounds__(256) k_res_gc1(
    const float* __restrict__ X,
    const float* __restrict__ rpw, const float* __restrict__ rpb,
    const float* __restrict__ w1, const float* __restrict__ b1,
    const float* __restrict__ w2, const float* __restrict__ b2,
    const float* __restrict__ w3, const float* __restrict__ b3,
    float* __restrict__ resid, float* __restrict__ g1out,
    float* __restrict__ bn0sums)
{
  __shared__ float wr[1024];         // rp_w [co][ci]
  __shared__ float wg[3][3168];      // gated weights, stride 99 (bank-conflict pad)
  __shared__ float xsm[320];         // [ci][t+1] with halo, stride 10
  __shared__ float bns[64];          // bn0 sum / sumsq
  const int tid = threadIdx.x;

  for (int i = tid; i < 1024; i += 256) wr[i] = rpw[i];
  const float* gw[3] = { w1, w2, w3 };
  for (int j = 0; j < 3; ++j)
    for (int i = tid; i < 3072; i += 256) {
      int cow = i / 96, k = i % 96;
      wg[j][cow * 99 + k] = gw[j][i];
    }
  if (tid < 64) bns[tid] = 0.0f;
  if (tid < 64) { int c = tid & 31, side = tid >> 5; xsm[c * 10 + (side ? 9 : 0)] = 0.0f; }

  const int co = tid >> 3, tt = tid & 7;
  const float rb = rpb[co], bb1 = b1[co], bb2 = b2[co], bb3 = b3[co];

  for (int nd = 0; nd < 16; ++nd) {
    const int n = blockIdx.x * 16 + nd;
    __syncthreads();
    xsm[(tid >> 3) * 10 + (tid & 7) + 1] = X[n * 256 + tid];
    __syncthreads();
    float r = rb, a1 = bb1, a2 = bb2, a3 = bb3;
#pragma unroll 4
    for (int ci = 0; ci < 32; ++ci) {
      float xm = xsm[ci * 10 + tt];
      float x0 = xsm[ci * 10 + tt + 1];
      float xp = xsm[ci * 10 + tt + 2];
      r += wr[co * 32 + ci] * x0;
      int wb_ = co * 99 + ci * 3;
      a1 += wg[0][wb_] * xm + wg[0][wb_ + 1] * x0 + wg[0][wb_ + 2] * xp;
      a2 += wg[1][wb_] * xm + wg[1][wb_ + 1] * x0 + wg[1][wb_ + 2] * xp;
      a3 += wg[2][wb_] * xm + wg[2][wb_ + 1] * x0 + wg[2][wb_ + 2] * xp;
    }
    float g = reluf_(a1 * sigmoidf_(a2) + a3);
    resid[n * 256 + tid] = r;
    g1out[n * 256 + tid] = g;
    atomicAdd(&bns[co], g);
    atomicAdd(&bns[32 + co], g * g);
  }
  __syncthreads();
  if (tid < 64) atomicAdd(&bn0sums[tid], bns[tid]);
}

// ---------------- BN finalize: a = g*rsqrt(var+eps), s = b - mean*a ----------------
__global__ void k_bn_finalize(const float* __restrict__ sums, const float* __restrict__ g,
                              const float* __restrict__ b, float* __restrict__ a,
                              float* __restrict__ s, int C, float inv_count)
{
  int i = blockIdx.x * blockDim.x + threadIdx.x;
  if (i >= C) return;
  float mean = sums[i] * inv_count;
  float var = sums[C + i] * inv_count - mean * mean;
  float av = g[i] * rsqrtf(var + BN_EPS);
  a[i] = av;
  s[i] = b[i] - mean * av;
}

// ---------------- apply bn0 + fp32 -> bf16 for GEMM A ----------------
__global__ void k_bn0_to_bf16(const float* __restrict__ g1, const float* __restrict__ a,
                              const float* __restrict__ s, unsigned short* __restrict__ xb, int total)
{
  int i = blockIdx.x * blockDim.x + threadIdx.x;
  if (i >= total) return;
  int c = (i >> 3) & 31;               // flat [C][T] -> channel
  xb[i] = f2bf(g1[i] * a[c] + s[c]);
}
__global__ void k_f32_to_bf16(const float* __restrict__ in, unsigned short* __restrict__ out, int total)
{
  int i = blockIdx.x * blockDim.x + threadIdx.x;
  if (i < total) out[i] = f2bf(in[i]);
}

// ---------------- K4: WMMA GEMM  h[N,1024] = xb[N,256] @ W[1024,256]^T  (bf16 -> f32) ----------------
// Block owns one 16-row M tile. Wave 0 TDM-loads the A tile into LDS (padded rows: 512B data +
// 16B pad -> 528B stride so ds_load_b128 A-fragment reads are bank-staggered). Each of the 8 waves
// covers two 64-column slabs; B fragments are double-buffered registers streamed from L2.
__global__ void __launch_bounds__(256) k_wmma_gemm(
    const unsigned short* __restrict__ xb, const unsigned short* __restrict__ wb,
    float* __restrict__ h)
{
  __shared__ __align__(16) unsigned char atile[16 * 528];   // 16 rows x (512B + 16B pad)
  const int tid  = threadIdx.x;
  const int lane = tid & 31;
  const int wave = tid >> 5;
  const int mi   = blockIdx.x;        // M tile index
  const int lh   = lane & 15;
  const int hi   = lane >> 4;

  if (tid < 32) {
    // ---- Tensor Data Mover: stage A tile (16 x 256 bf16) global -> LDS ----
    unsigned long long ga = (unsigned long long)(uintptr_t)(xb + (size_t)mi * 4096);
    unsigned lds = (unsigned)(uintptr_t)(void*)atile;
    v4u g0;
    g0[0] = 1u;                                               // count=1, user descriptor
    g0[1] = lds;                                              // lds_addr (bytes)
    g0[2] = (unsigned)(ga & 0xFFFFFFFFu);                     // global_addr[31:0]
    g0[3] = ((unsigned)((ga >> 32) & 0x01FFFFFFu)) | 0x80000000u; // global_addr[56:32] | type=2
    v8i_t g1d;
    // data_size=2B (code 1) | pad_enable | pad_interval=6 (128 DWORDs) | pad_amount=3 (4 DWORDs)
    g1d[0] = (int)((1u << 16) | (1u << 20) | (6u << 22) | (3u << 25));
    g1d[1] = (int)(256u << 16);        // tensor_dim0 = 256 (bits 79:48, low half)
    g1d[2] = (int)(16u << 16);         // tensor_dim0 hi = 0; tensor_dim1 = 16 (bits 111:80, low half)
    g1d[3] = (int)(256u << 16);        // tensor_dim1 hi = 0; tile_dim0 = 256 (bits 127:112)
    g1d[4] = (int)16;                  // tile_dim1 = 16; tile_dim2 = 0
    g1d[5] = (int)256;                 // tensor_dim0_stride = 256 elements
    g1d[6] = 0;                        // stride hi / tensor_dim1_stride lo
    g1d[7] = 0;
    v4i_t zz4 = { 0, 0, 0, 0 };
    v8i_t zz8 = { 0, 0, 0, 0, 0, 0, 0, 0 };
    __builtin_amdgcn_tensor_load_to_lds(g0, g1d, zz4, zz4, zz8, 0);
    __builtin_amdgcn_s_wait_tensorcnt(0);
  }
  __syncthreads();

  const uint4* ap = (const uint4*)(atile + lh * 528);
  const unsigned* wbase = (const unsigned*)wb;

#pragma unroll
  for (int pass = 0; pass < 2; ++pass) {
    const int ni = wave + pass * 8;    // 64-column slab index (0..15)
    v8f acc[4];
#pragma unroll
    for (int s = 0; s < 4; ++s) acc[s] = (v8f){0.f,0.f,0.f,0.f,0.f,0.f,0.f,0.f};

    union BV { v16bf v; uint4 q[2]; };
    BV bbuf[2][4];
    // prime kk=0 B fragments: col = ni*64 + sub*16 + lh, 16 contiguous halves from W row
#pragma unroll
    for (int sub = 0; sub < 4; ++sub) {
      const uint4* pb = (const uint4*)(wbase + (ni * 64 + sub * 16 + lh) * 128 + hi * 8);
      bbuf[0][sub].q[0] = pb[0];
      bbuf[0][sub].q[1] = pb[1];
    }
#pragma unroll
    for (int kk = 0; kk < 8; ++kk) {
      const int cur = kk & 1, nxt = cur ^ 1;
      if (kk < 7) {
#pragma unroll
        for (int sub = 0; sub < 4; ++sub) {
          const uint4* pb = (const uint4*)(wbase + (ni * 64 + sub * 16 + lh) * 128 +
                                           (kk + 1) * 16 + hi * 8);
          bbuf[nxt][sub].q[0] = pb[0];
          bbuf[nxt][sub].q[1] = pb[1];
        }
      }
      union { v16bf v; uint4 q[2]; } A;     // A fragment from LDS (contiguous k = 16*hi + s, same as B)
      A.q[0] = ap[kk * 4 + 2 * hi];
      A.q[1] = ap[kk * 4 + 2 * hi + 1];
#pragma unroll
      for (int sub = 0; sub < 4; ++sub) {
        acc[sub] = __builtin_amdgcn_wmma_f32_16x16x32_bf16(
            false, A.v, false, bbuf[cur][sub].v, (short)0, acc[sub], false, false);
      }
    }
    // C/D layout: VGPR r, lane l -> (M = r + 8*(l/16), N = l%16)
#pragma unroll
    for (int sub = 0; sub < 4; ++sub) {
      float* hp = h + (mi * 16 + hi * 8) * 1024 + ni * 64 + sub * 16 + lh;
#pragma unroll
      for (int r = 0; r < 8; ++r) hp[r * 1024] = acc[sub][r];
    }
  }
}

// ---------------- K5: el/er = sum_f h[n,h,f] * attn_{l,r}[h,f]  (one wave per (n,head)) ----------------
__global__ void k_elr(const float* __restrict__ h, const float* __restrict__ al,
                      const float* __restrict__ ar, float* __restrict__ el,
                      float* __restrict__ er, int N)
{
  int wv = (blockIdx.x * blockDim.x + threadIdx.x) >> 5;
  int lane = threadIdx.x & 31;
  if (wv >= N * 4) return;
  int n = wv >> 2, hh = wv & 3;
  const float* hr  = h + n * 1024 + hh * 256;
  const float* alr = al + hh * 256;
  const float* arr = ar + hh * 256;
  float sl = 0.f, sr = 0.f;
#pragma unroll
  for (int i = 0; i < 8; ++i) {
    float v = hr[lane + i * 32];
    sl += v * alr[lane + i * 32];
    sr += v * arr[lane + i * 32];
  }
#pragma unroll
  for (int off = 16; off > 0; off >>= 1) {
    sl += __shfl_down(sl, off, 32);
    sr += __shfl_down(sr, off, 32);
  }
  if (lane == 0) { el[wv] = sl; er[wv] = sr; }
}

// ---------------- K6/K7: edge softmax (segment max via monotone-uint atomicMax, then exp+sum) ------
__global__ void k_edge_max(const int* __restrict__ src, const int* __restrict__ dst,
                           const float* __restrict__ el, const float* __restrict__ er,
                           unsigned* __restrict__ mx, int EH)
{
  int i = blockIdx.x * blockDim.x + threadIdx.x;
  if (i >= EH) return;
  int e = i >> 2, hh = i & 3;
  int s = src[e], d = dst[e];
  float x = el[s * 4 + hh] + er[d * 4 + hh];
  x = x > 0.f ? x : NEG_SLOPE * x;
  atomicMax(&mx[d * 4 + hh], fenc(x));
}
__global__ void k_edge_exp(const int* __restrict__ src, const int* __restrict__ dst,
                           const float* __restrict__ el, const float* __restrict__ er,
                           const unsigned* __restrict__ mx, float* __restrict__ ee,
                           float* __restrict__ den, int EH)
{
  int i = blockIdx.x * blockDim.x + threadIdx.x;
  if (i >= EH) return;
  int e = i >> 2, hh = i & 3;
  int s = src[e], d = dst[e];
  float x = el[s * 4 + hh] + er[d * 4 + hh];
  x = x > 0.f ? x : NEG_SLOPE * x;
  float v = __expf(x - fdec(mx[d * 4 + hh]));
  ee[i] = v;
  atomicAdd(&den[d * 4 + hh], v);
}

// ---------------- K8: rst[dst] += alpha * h[src]  (one block per edge; h lives in L2) -------------
__global__ void __launch_bounds__(256) k_aggregate(
    const int* __restrict__ src, const int* __restrict__ dst,
    const float* __restrict__ ee, const float* __restrict__ den,
    const float* __restrict__ h, float* __restrict__ rst)
{
  const int e = blockIdx.x;
  const int tid = threadIdx.x;
  const int s = src[e], d = dst[e];
  const float* hs = h + s * 1024;
  float* rd = rst + d * 1024;
#pragma unroll
  for (int i = 0; i < 4; ++i) {
    float a = ee[e * 4 + i] / den[d * 4 + i];
    int idx = i * 256 + tid;
    atomicAdd(&rd[idx], a * hs[idx]);
  }
}

// ---------------- K9: bn1 column stats of relu(rst + gat_bias) over N ----------------
__global__ void k_bn1_stats(const float* __restrict__ rst, const float* __restrict__ bias,
                            float* __restrict__ sums, int N)
{
  __shared__ float red[512];
  const int tid = threadIdx.x;
  const int cl = tid & 31;
  const int r0 = tid >> 5;
  const int col = blockIdx.x * 32 + cl;
  const float b = bias[col];
  float s = 0.f, s2 = 0.f;
  for (int n = r0; n < N; n += 8) {
    float v = reluf_(rst[n * 1024 + col] + b);
    s += v; s2 += v * v;
  }
  red[tid] = s; red[256 + tid] = s2;
  __syncthreads();
  if (r0 == 0) {
    float ts = 0.f, t2 = 0.f;
#pragma unroll
    for (int k = 0; k < 8; ++k) { ts += red[k * 32 + cl]; t2 += red[256 + k * 32 + cl]; }
    sums[col] = ts;
    sums[1024 + col] = t2;
  }
}

// ---------------- K11: bn1-apply + reshape + gated_conv2 (128->32, k=3) + bn2 stats ---------------
__global__ void __launch_bounds__(256) k_gconv2(
    const float* __restrict__ rst, const float* __restrict__ gbias,
    const float* __restrict__ a1, const float* __restrict__ s1,
    const float* __restrict__ w1, const float* __restrict__ b1,
    const float* __restrict__ w2, const float* __restrict__ b2,
    const float* __restrict__ w3, const float* __restrict__ b3,
    float* __restrict__ g2out, float* __restrict__ bn2sums)
{
  __shared__ float xs[4][1280];     // 4 nodes, [ci][t+1] halo layout, stride 10
  __shared__ float wsm[3][3168];    // weight chunk (32 ci), stride 99
  __shared__ float bns[64];
  const int tid = threadIdx.x;
  if (tid < 64) bns[tid] = 0.f;
#pragma unroll
  for (int z = 0; z < 4; ++z) {     // zero halos
    int j = z * 256 + tid;
    int nd = j >> 8, rem = j & 255;
    int c = rem & 127, side = rem >> 7;
    xs[nd][c * 10 + (side ? 9 : 0)] = 0.f;
  }
  const int n0 = blockIdx.x * 4;
  for (int nd = 0; nd < 4; ++nd) {
#pragma unroll
    for (int i = 0; i < 4; ++i) {
      int col = i * 256 + tid;
      float v = reluf_(rst[(n0 + nd) * 1024 + col] + gbias[col]);
      v = v * a1[col] + s1[col];
      xs[nd][(col >> 3) * 10 + (col & 7) + 1] = v;
    }
  }
  const int co = tid >> 3, t = tid & 7;
  float acc[3][4];
  {
    float bb1 = b1[co], bb2 = b2[co], bb3 = b3[co];
    for (int nd = 0; nd < 4; ++nd) { acc[0][nd] = bb1; acc[1][nd] = bb2; acc[2][nd] = bb3; }
  }
  const float* gw[3] = { w1, w2, w3 };
  for (int chunk = 0; chunk < 4; ++chunk) {
    const int ci0 = chunk * 32;
    __syncthreads();
    for (int j = 0; j < 3; ++j)
#pragma unroll
      for (int q = 0; q < 12; ++q) {
        int idx = q * 256 + tid;
        int cow = idx / 96, k = idx % 96;
        wsm[j][cow * 99 + k] = gw[j][cow * 384 + ci0 * 3 + k];
      }
    __syncthreads();
    for (int nd = 0; nd < 4; ++nd) {
#pragma unroll 4
      for (int cl = 0; cl < 32; ++cl) {
        int ci = ci0 + cl;
        float xm = xs[nd][ci * 10 + t];
        float x0 = xs[nd][ci * 10 + t + 1];
        float xp = xs[nd][ci * 10 + t + 2];
        int wb_ = co * 99 + cl * 3;
        acc[0][nd] += wsm[0][wb_] * xm + wsm[0][wb_ + 1] * x0 + wsm[0][wb_ + 2] * xp;
        acc[1][nd] += wsm[1][wb_] * xm + wsm[1][wb_ + 1] * x0 + wsm[1][wb_ + 2] * xp;
        acc[2][nd] += wsm[2][wb_] * xm + wsm[2][wb_ + 1] * x0 + wsm[2][wb_ + 2] * xp;
      }
    }
  }
  for (int nd = 0; nd < 4; ++nd) {
    float g = reluf_(acc[0][nd] * sigmoidf_(acc[1][nd]) + acc[2][nd]);
    g2out[(n0 + nd) * 256 + tid] = g;
    atomicAdd(&bns[co], g);
    atomicAdd(&bns[32 + co], g * g);
  }
  __syncthreads();
  if (tid < 64) atomicAdd(&bn2sums[tid], bns[tid]);
}

// ---------------- K13: out = relu(bn2(g2) + residual) ----------------
__global__ void k_final(const float* __restrict__ g2, const float* __restrict__ resid,
                        const float* __restrict__ a, const float* __restrict__ s,
                        float* __restrict__ out, int total)
{
  int i = blockIdx.x * blockDim.x + threadIdx.x;
  if (i >= total) return;
  int c = (i >> 3) & 31;
  float v = g2[i] * a[c] + s[c] + resid[i];
  out[i] = reluf_(v);
}

extern "C" void kernel_launch(void* const* d_in, const int* in_sizes, int n_in,
                              void* d_out, int out_size, void* d_ws, size_t ws_size,
                              hipStream_t stream)
{
  (void)n_in; (void)out_size; (void)ws_size;
  const float* X    = (const float*)d_in[0];
  const int*   src  = (const int*)d_in[1];
  const int*   dst  = (const int*)d_in[2];
  const float* rp_w = (const float*)d_in[3];
  const float* rp_b = (const float*)d_in[4];
  const float* g1w1 = (const float*)d_in[5];
  const float* g1b1 = (const float*)d_in[6];
  const float* g1w2 = (const float*)d_in[7];
  const float* g1b2 = (const float*)d_in[8];
  const float* g1w3 = (const float*)d_in[9];
  const float* g1b3 = (const float*)d_in[10];
  const float* bn0g = (const float*)d_in[11];
  const float* bn0b = (const float*)d_in[12];
  const float* gatw = (const float*)d_in[13];
  const float* attl = (const float*)d_in[14];
  const float* attr = (const float*)d_in[15];
  const float* gatb = (const float*)d_in[16];
  const float* bn1g = (const float*)d_in[17];
  const float* bn1b = (const float*)d_in[18];
  const float* g2w1 = (const float*)d_in[19];
  const float* g2b1 = (const float*)d_in[20];
  const float* g2w2 = (const float*)d_in[21];
  const float* g2b2 = (const float*)d_in[22];
  const float* g2w3 = (const float*)d_in[23];
  const float* g2b3 = (const float*)d_in[24];
  const float* bn2g = (const float*)d_in[25];
  const float* bn2b = (const float*)d_in[26];

  const int N = in_sizes[0] / 256;    // 20000
  const int E = in_sizes[1];          // 320000
  const int NF = N * 256, NHF = N * 1024;

  char* base = (char*)d_ws;
  size_t off = 0;
  auto alloc = [&](size_t bytes) -> char* {
    char* p = base + off;
    off += (bytes + 255) & ~(size_t)255;
    return p;
  };
  float* resid = (float*)alloc(sizeof(float) * NF);
  float* hbuf  = (float*)alloc(sizeof(float) * NHF);
  float* rstb  = (float*)alloc(sizeof(float) * NHF);
  unsigned short* xb  = (unsigned short*)alloc(sizeof(unsigned short) * NF);
  unsigned short* wbv = (unsigned short*)alloc(sizeof(unsigned short) * 1024 * 256);
  float* el  = (float*)alloc(sizeof(float) * N * 4);
  float* er  = (float*)alloc(sizeof(float) * N * 4);
  float* den = (float*)alloc(sizeof(float) * N * 4);
  unsigned* mx = (unsigned*)alloc(sizeof(unsigned) * N * 4);
  float* ee  = (float*)alloc(sizeof(float) * (size_t)E * 4);
  float* stats = (float*)alloc(sizeof(float) * 4352);

  float* g1out = rstb;   // alias: consumed before rst is zeroed
  float* g2out = hbuf;   // alias: h dead after aggregation

  float* bn0sums = stats;            // [0..63]   (sum 0-31, sumsq 32-63)
  float* bn1sums = stats + 64;       // [64..2111]
  float* bn2sums = stats + 2112;     // [2112..2175]
  float* bn0a = stats + 2176; float* bn0s = stats + 2208;
  float* bn1a = stats + 2240; float* bn1s = stats + 3264;
  float* bn2a = stats + 4288; float* bn2s = stats + 4320;

  (void)hipMemsetAsync(stats, 0, sizeof(float) * 2176, stream);
  (void)hipMemsetAsync(mx, 0, sizeof(unsigned) * N * 4, stream);
  (void)hipMemsetAsync(den, 0, sizeof(float) * N * 4, stream);

  k_res_gc1<<<N / 16, 256, 0, stream>>>(X, rp_w, rp_b, g1w1, g1b1, g1w2, g1b2, g1w3, g1b3,
                                        resid, g1out, bn0sums);
  k_bn_finalize<<<1, 32, 0, stream>>>(bn0sums, bn0g, bn0b, bn0a, bn0s, 32, 1.0f / (N * 8.0f));
  k_bn0_to_bf16<<<(NF + 255) / 256, 256, 0, stream>>>(g1out, bn0a, bn0s, xb, NF);
  k_f32_to_bf16<<<(262144 + 255) / 256, 256, 0, stream>>>(gatw, wbv, 262144);
  (void)hipMemsetAsync(rstb, 0, sizeof(float) * NHF, stream);
  k_wmma_gemm<<<N / 16, 256, 0, stream>>>(xb, wbv, hbuf);
  k_elr<<<(N * 4 + 7) / 8, 256, 0, stream>>>(hbuf, attl, attr, el, er, N);
  k_edge_max<<<(E * 4 + 255) / 256, 256, 0, stream>>>(src, dst, el, er, mx, E * 4);
  k_edge_exp<<<(E * 4 + 255) / 256, 256, 0, stream>>>(src, dst, el, er, mx, ee, den, E * 4);
  k_aggregate<<<E, 256, 0, stream>>>(src, dst, ee, den, hbuf, rstb);
  k_bn1_stats<<<32, 256, 0, stream>>>(rstb, gatb, bn1sums, N);
  k_bn_finalize<<<4, 256, 0, stream>>>(bn1sums, bn1g, bn1b, bn1a, bn1s, 1024, 1.0f / N);
  k_gconv2<<<N / 4, 256, 0, stream>>>(rstb, gatb, bn1a, bn1s,
                                      g2w1, g2b1, g2w2, g2b2, g2w3, g2b3, g2out, bn2sums);
  k_bn_finalize<<<1, 32, 0, stream>>>(bn2sums, bn2g, bn2b, bn2a, bn2s, 32, 1.0f / (N * 8.0f));
  k_final<<<(NF + 255) / 256, 256, 0, stream>>>(g2out, resid, bn2a, bn2s, (float*)d_out, NF);
}
